// RNN_ConcatTime_14783277433333
// MI455X (gfx1250) — matrix operand order repaired and checked by reading, not compiled, not run on hardware
//
#include <hip/hip_runtime.h>
#include <hip/hip_bf16.h>

#define B_ 4096
#define S_ 512
#define HD_ 9
#define HC_ 11

typedef _Float16 v16h __attribute__((ext_vector_type(16)));
typedef float    v8f  __attribute__((ext_vector_type(8)));

#define LOG2E_ 1.4426950408889634f

__device__ __forceinline__ float fast_sigmoid(float x) {
    // rcp(1 + exp2(-x*log2e)) : 2 TRANS + 2 VALU, no IEEE division sequence
    return __builtin_amdgcn_rcpf(1.f + __builtin_amdgcn_exp2f(-LOG2E_ * x));
}

__device__ __forceinline__ float fast_tanh(float x) {
#if __has_builtin(__builtin_amdgcn_tanhf)
    return __builtin_amdgcn_tanhf(x);          // native v_tanh_f32 on gfx1250
#else
    // branchless: tanh(x) = 1 - 2/(exp(2x)+1); saturates correctly at +-inf
    const float e = __builtin_amdgcn_exp2f(2.f * LOG2E_ * x);
    return 1.f - 2.f * __builtin_amdgcn_rcpf(e + 1.f);
#endif
}

// One wave per block; each wave owns a 16-row batch tile and runs the full
// 512-step recurrence with 4x WMMA f32_16x16x32_f16 per step.
//
// K-fusion: A = [x | h] stacked along K (x -> K 0..15, h -> K 16..31).
//   A-layout maps this perfectly: lane<16 holds K 0..7 (elems 0..7) and
//   K 16..23 (elems 8..15); lane>=16 holds K 8..15 and K 24..31. So with
//   kbase = lane<16 ? 0 : 8:  a[e] = x[kbase+e],  a[8+j] = h[kbase+j].
// B tiles (N = hidden unit j, padded per-gate to 16 cols):
//   rz[g] (g=0:r, 1:z): K 0..15 rows = Wih[g], K 16..31 rows = Whh[g]
//                       -> one WMMA yields x*Wih + h*Whh (+ bih + bhh in C).
//   xn: K 0..15 = Wih[2] only;  hn: K 16..31 = Whh[2] only (r gates hn).
// C-layout: lane<16 -> col N=lane, elem k = row M=k; lane>=16 -> col N=lane-16,
//           elem k = row M=8+k.  Padding cols stay exactly 0 through the gates.
//
// The embedding gather is *arithmetically* selected (precomputed 0/1 float
// masks) so the loaded value is unconditionally consumed: the compiler cannot
// sink the loads under lane-divergent predicates, keeping the loop body
// straight-line (no saveexec blocks, no per-element xcnt waits).
template<int ED, int H, bool FWD>
__global__ __launch_bounds__(32)
void gru_wmma_kernel(const int* __restrict__ idx, const float* __restrict__ tvec,
                     const float* __restrict__ emb,
                     const float* __restrict__ Wih, const float* __restrict__ Whh,
                     const float* __restrict__ bih, const float* __restrict__ bhh,
                     float* __restrict__ outH)
{
    const int  lane  = threadIdx.x;       // 0..31
    const int  base  = blockIdx.x * 16;   // batch tile start
    const int  col   = lane & 15;         // owned column (N for B/C, M for A)
    const bool lo    = lane < 16;
    const int  mrow0 = lo ? 0 : 8;        // first C-layout row this lane owns
    const int  kbase = lo ? 0 : 8;        // first K (within each 16-half) held

    __shared__ _Float16 hbuf[16 * 16];    // hidden state, [m][j], zero-padded

    // ---- constant B fragments and bias C-initializer fragments
    v16h wB[4];          // 0:r (ih|hh), 1:z (ih|hh), 2:xn (ih only), 3:hn (hh only)
    v8f  cB[4];
#pragma unroll
    for (int g = 0; g < 2; ++g) {         // fused r,z tiles
        v16h w = {};
        float bb = 0.f;
        if (col < H) {
            const int row = g * H + col;
            const float* Wp = Wih + row * H;
            const float* Up = Whh + row * H;
#pragma unroll
            for (int e = 0; e < 16; ++e) {
                const int kk = kbase + (e & 7);
                if (kk < H) w[e] = (_Float16)((e < 8) ? Wp[kk] : Up[kk]);
            }
            bb = bih[row] + bhh[row];
        }
        wB[g] = w;
        v8f c;
#pragma unroll
        for (int k = 0; k < 8; ++k) c[k] = bb;
        cB[g] = c;
    }
    {                                      // xn / hn tiles
        v16h wx = {}, wh = {};
        float bx = 0.f, bh = 0.f;
        if (col < H) {
            const int row = 2 * H + col;
#pragma unroll
            for (int e = 0; e < 16; ++e) {
                const int kk = kbase + (e & 7);
                if (kk < H) {
                    if (e < 8) wx[e] = (_Float16)Wih[row * H + kk];   // K 0..15 (elems 0..7)
                    else       wh[e] = (_Float16)Whh[row * H + kk];   // K 16..31 (elems 8..15)
                }
            }
            bx = bih[row]; bh = bhh[row];
        }
        wB[2] = wx; wB[3] = wh;
        v8f cx, ch;
#pragma unroll
        for (int k = 0; k < 8; ++k) { cx[k] = bx; ch[k] = bh; }
        cB[2] = cx; cB[3] = ch;
    }

    // ---- per-lane gather constants: clamped emb offsets + 0/1 select masks
    int   kcl[8];                          // clamped emb column (always in-bounds)
    float mEmb[8], mDt[8];                 // 1.0 where element is emb / delta
#pragma unroll
    for (int e = 0; e < 8; ++e) {
        const int k = kbase + e;
        kcl[e]  = (k < ED) ? k : 0;
        mEmb[e] = (k < ED) ? 1.f : 0.f;
        mDt[e]  = (k == ED) ? 1.f : 0.f;
    }

    // ---- init hidden state (registers + LDS copy)
    v8f hC = {};
    for (int i = lane; i < 256; i += 32) hbuf[i] = (_Float16)0.f;
    __syncthreads();

    const int brow = base + col;          // batch row this lane gathers/owns

    // ---- prologue: gather x for step 0
    float t_prev = 0.f;
    float xv[8];                          // current-step x features (f32)
    {
        const int   so = FWD ? 0 : (S_ - 1);
        const int   id = idx[brow * S_ + so];
        const float tc = tvec[brow * S_ + so];
        const float dt = tc > 0.f ? tc : 0.f;   // t_prev = 0
        t_prev = tc;
        const float* er = emb + (size_t)id * ED;
#pragma unroll
        for (int e = 0; e < 8; ++e)
            xv[e] = er[kcl[e]] * mEmb[e] + dt * mDt[e];
    }

    for (int s = 0; s < S_; ++s) {
        // ---- issue next-step idx/time loads early (clamped step: branch-free;
        //      last iteration redoes the final step's gather, result unused)
        const int   s1   = (s + 1 < S_) ? (s + 1) : (S_ - 1);
        const int   so1  = FWD ? s1 : (S_ - 1 - s1);
        const int   id_n = idx[brow * S_ + so1];
        const float tc_n = tvec[brow * S_ + so1];

        // ---- combined A = [x | h]: x in elems 0..7, h (from LDS) in 8..15
        v16h a;
#pragma unroll
        for (int e = 0; e < 8; ++e) a[e] = (_Float16)xv[e];
        {
            const _Float16* hp = &hbuf[col * 16 + kbase];
#pragma unroll
            for (int j = 0; j < 8; ++j) a[8 + j] = hp[j];
        }

        // ---- 4x WMMA (r and z fully fused; n split into xn / hn)
        v8f rz0 = __builtin_amdgcn_wmma_f32_16x16x32_f16(false, a, false, wB[0], (short)0, cB[0], false, false);
        v8f rz1 = __builtin_amdgcn_wmma_f32_16x16x32_f16(false, a, false, wB[1], (short)0, cB[1], false, false);
        v8f xn  = __builtin_amdgcn_wmma_f32_16x16x32_f16(false, a, false, wB[2], (short)0, cB[2], false, false);
        v8f hn  = __builtin_amdgcn_wmma_f32_16x16x32_f16(false, a, false, wB[3], (short)0, cB[3], false, false);

        // ---- prefetch next-step embedding row (overlaps with gate math);
        //      mask-multiply select: loads are unconditionally consumed, so the
        //      compiler keeps them straight-line (no exec-masked blocks)
        float dt_n = tc_n - t_prev;
        dt_n = dt_n > 0.f ? dt_n : 0.f;
        t_prev = tc_n;
        {
            const float* er = emb + (size_t)id_n * ED;
#pragma unroll
            for (int e = 0; e < 8; ++e)
                xv[e] = er[kcl[e]] * mEmb[e] + dt_n * mDt[e];
        }

        // ---- gates (per-lane elementwise thanks to per-gate column padding)
#pragma unroll
        for (int k = 0; k < 8; ++k) {
            const float r = fast_sigmoid(rz0[k]);
            const float z = fast_sigmoid(rz1[k]);
            const float n = fast_tanh(xn[k] + r * hn[k]);
            hC[k] = (1.f - z) * n + z * hC[k];
        }

        __syncthreads();   // single-wave WG: barrier is free; orders LDS RAW/WAR
#pragma unroll
        for (int k = 0; k < 8; ++k)
            hbuf[(mrow0 + k) * 16 + col] = (_Float16)hC[k];
        __syncthreads();
    }

    // ---- write final hidden state h_T (B x H) to workspace
    if (col < H) {
#pragma unroll
        for (int k = 0; k < 8; ++k)
            outH[(size_t)(base + mrow0 + k) * H + col] = hC[k];
    }
}

// Final dense combine: score_dp, score_cp, then fc_all. One thread per batch row.
__global__ __launch_bounds__(256)
void combine_kernel(const float* __restrict__ stat,
                    const float* __restrict__ hdf, const float* __restrict__ hdb,
                    const float* __restrict__ hcf, const float* __restrict__ hcb,
                    const float* __restrict__ wdp, const float* __restrict__ bdp,
                    const float* __restrict__ wcp, const float* __restrict__ bcp,
                    const float* __restrict__ wall, const float* __restrict__ ball,
                    float* __restrict__ out)
{
    const int b = blockIdx.x * blockDim.x + threadIdx.x;
    if (b >= B_) return;
    float sdp = bdp[0];
#pragma unroll
    for (int j = 0; j < HD_; ++j)
        sdp += hdf[b * HD_ + j] * wdp[j] + hdb[b * HD_ + j] * wdp[HD_ + j];
    float scp = bcp[0];
#pragma unroll
    for (int j = 0; j < HC_; ++j)
        scp += hcf[b * HC_ + j] * wcp[j] + hcb[b * HC_ + j] * wcp[HC_ + j];
    float o = ball[0];
#pragma unroll
    for (int i = 0; i < 20; ++i)
        o += stat[b * 20 + i] * wall[i];
    o += sdp * wall[20] + scp * wall[21];
    out[b] = o;
}

extern "C" void kernel_launch(void* const* d_in, const int* in_sizes, int n_in,
                              void* d_out, int out_size, void* d_ws, size_t ws_size,
                              hipStream_t stream) {
    (void)in_sizes; (void)n_in; (void)out_size; (void)ws_size;

    const float* stat  = (const float*)d_in[0];
    const int*   dp    = (const int*)  d_in[1];
    const int*   cp    = (const int*)  d_in[2];
    const float* dp_t  = (const float*)d_in[3];
    const float* cp_t  = (const float*)d_in[4];
    const float* edp   = (const float*)d_in[5];
    const float* ecp   = (const float*)d_in[6];
    // dpf: 7..10, dpb: 11..14, cpf: 15..18, cpb: 19..22
    const float* Wih_dpf = (const float*)d_in[7],  *Whh_dpf = (const float*)d_in[8];
    const float* bih_dpf = (const float*)d_in[9],  *bhh_dpf = (const float*)d_in[10];
    const float* Wih_dpb = (const float*)d_in[11], *Whh_dpb = (const float*)d_in[12];
    const float* bih_dpb = (const float*)d_in[13], *bhh_dpb = (const float*)d_in[14];
    const float* Wih_cpf = (const float*)d_in[15], *Whh_cpf = (const float*)d_in[16];
    const float* bih_cpf = (const float*)d_in[17], *bhh_cpf = (const float*)d_in[18];
    const float* Wih_cpb = (const float*)d_in[19], *Whh_cpb = (const float*)d_in[20];
    const float* bih_cpb = (const float*)d_in[21], *bhh_cpb = (const float*)d_in[22];
    const float* fc_dp_w = (const float*)d_in[23], *fc_dp_b = (const float*)d_in[24];
    const float* fc_cp_w = (const float*)d_in[25], *fc_cp_b = (const float*)d_in[26];
    const float* fc_all_w= (const float*)d_in[27], *fc_all_b= (const float*)d_in[28];

    float* ws   = (float*)d_ws;
    float* hdpf = ws;                       // B*9
    float* hdpb = hdpf + (size_t)B_ * HD_;  // B*9
    float* hcpf = hdpb + (size_t)B_ * HD_;  // B*11
    float* hcpb = hcpf + (size_t)B_ * HC_;  // B*11

    const dim3 grid(B_ / 16), block(32);
    gru_wmma_kernel<8, HD_, true ><<<grid, block, 0, stream>>>(dp, dp_t, edp, Wih_dpf, Whh_dpf, bih_dpf, bhh_dpf, hdpf);
    gru_wmma_kernel<8, HD_, false><<<grid, block, 0, stream>>>(dp, dp_t, edp, Wih_dpb, Whh_dpb, bih_dpb, bhh_dpb, hdpb);
    gru_wmma_kernel<10, HC_, true ><<<grid, block, 0, stream>>>(cp, cp_t, ecp, Wih_cpf, Whh_cpf, bih_cpf, bhh_cpf, hcpf);
    gru_wmma_kernel<10, HC_, false><<<grid, block, 0, stream>>>(cp, cp_t, ecp, Wih_cpb, Whh_cpb, bih_cpb, bhh_cpb, hcpb);

    combine_kernel<<<(B_ + 255) / 256, 256, 0, stream>>>(
        stat, hdpf, hdpb, hcpf, hcpb,
        fc_dp_w, fc_dp_b, fc_cp_w, fc_cp_b, fc_all_w, fc_all_b,
        (float*)d_out);
}
